// Net_13262859010241
// MI455X (gfx1250) — compile-verified, matrix operand order checked by reading, not run on hardware
//
#include <hip/hip_runtime.h>
#include <hip/hip_bf16.h>
#include <math.h>

typedef __attribute__((ext_vector_type(16))) _Float16 v16h;
typedef __attribute__((ext_vector_type(8)))  float    v8f;

#define BATCH  512
#define SEQLEN 500
#define ROWS   (BATCH*SEQLEN)   // 256000
#define MEM    64
#define DK     128
#define DV     256
#define FCN    128
#define KIN    (DV+DK)          // 384

// ---- WMMA helpers -----------------------------------------------------------
// 16-bit A/B fragment per CDNA5 ISA 7.12.2 (16x32 f16):
// lane L(<16): M=L, VGPR0..3 = K 0..7, VGPR4..7 = K 16..23
// lane L+16 :  M=L, VGPR0..3 = K 8..15, VGPR4..7 = K 24..31
// B uses the same per-lane K striping with lane&15 selecting column N.
__device__ __forceinline__ v16h frag16(const _Float16* base, int ld, int lane, int k0) {
  v16h a;
  const int hi = (lane >> 4) & 1;
  const _Float16* r = base + (lane & 15) * ld;
#pragma unroll
  for (int j = 0; j < 8; ++j) {
    const int K = k0 + ((2 * j) & 7) + 8 * hi + 16 * (j >> 2);
    a[2 * j]     = r[K];
    a[2 * j + 1] = r[K + 1];
  }
  return a;
}

__device__ __forceinline__ v8f wmma16(v16h a, v16h b, v8f c) {
  return __builtin_amdgcn_wmma_f32_16x16x32_f16(false, a, false, b, (short)0, c,
                                                false, false);
}

__device__ __forceinline__ float sigf(float x) { return 1.f / (1.f + expf(-x)); }

// ---- Stage 1: cw = softmax_m(q_e . key_memory) ------------------------------
// 128 threads = 4 waves, 64 rows per workgroup (16 rows per wave).
__global__ __launch_bounds__(128)
void cw_kernel(const int* __restrict__ q_data, const float* __restrict__ q_embed,
               const float* __restrict__ key_mem, float* __restrict__ cw) {
  __shared__ _Float16 ldsQ[64 * DK];      // 16 KB: gathered q_e rows (f16)
  __shared__ _Float16 ldsK[MEM * DK];     // 16 KB: key memory (f16)
  __shared__ float    ldsL[64 * MEM];     // 16 KB: raw logits for softmax
  __shared__ int      idxS[64];
  const int tid = threadIdx.x, lane = tid & 31, w = tid >> 5;
  const long row0 = (long)blockIdx.x * 64;

  if (tid < 64) idxS[tid] = q_data[row0 + tid];
  for (int i = tid; i < MEM * DK; i += 128) ldsK[i] = (_Float16)key_mem[i];
  __syncthreads();
  for (int i = tid; i < 64 * DK; i += 128) {
    const int r = i >> 7, c = i & 127;
    ldsQ[i] = (_Float16)q_embed[(long)idxS[r] * DK + c];
  }
  __syncthreads();

  const _Float16* aB = ldsQ + w * 16 * DK;
  v16h af[4];
#pragma unroll
  for (int k = 0; k < 4; ++k) af[k] = frag16(aB, DK, lane, k * 32);

#pragma unroll
  for (int nt = 0; nt < 4; ++nt) {   // M = 64 -> 4 N-tiles
    v8f acc = {0.f, 0.f, 0.f, 0.f, 0.f, 0.f, 0.f, 0.f};
#pragma unroll
    for (int k = 0; k < 4; ++k)      // K = 128 -> 4 k-steps
      acc = wmma16(af[k], frag16(ldsK + nt * 16 * DK, DK, lane, k * 32), acc);
    const int col = nt * 16 + (lane & 15);
    const int rbase = w * 16 + 8 * (lane >> 4);
#pragma unroll
    for (int r = 0; r < 8; ++r) ldsL[(rbase + r) * MEM + col] = acc[r];
  }
  __syncthreads();

  if (tid < 64) {   // per-row softmax over M=64
    float mx = -1e30f;
#pragma unroll
    for (int j = 0; j < MEM; ++j) mx = fmaxf(mx, ldsL[tid * MEM + j]);
    float sum = 0.f;
#pragma unroll
    for (int j = 0; j < MEM; ++j) sum += expf(ldsL[tid * MEM + j] - mx);
    const float inv = 1.f / sum;
    float* o = cw + (row0 + tid) * MEM;
#pragma unroll
    for (int j = 0; j < MEM; ++j) o[j] = expf(ldsL[tid * MEM + j] - mx) * inv;
  }
}

// ---- Stage 2: erase = sigmoid(qa_e @ Ew^T + be), add = tanh(qa_e @ Aw^T + ba)
__global__ __launch_bounds__(128)
void ea_kernel(const int* __restrict__ qa_data, const float* __restrict__ qa_embed,
               const float* __restrict__ erase_W, const float* __restrict__ erase_b,
               const float* __restrict__ add_W, const float* __restrict__ add_b,
               _Float16* __restrict__ eraseO, _Float16* __restrict__ addO) {
  __shared__ _Float16 ldsQA[64 * DV];   // 32 KB gathered qa_e (f16)
  __shared__ _Float16 ldsWE[16 * DV];   // 8 KB erase_W tile
  __shared__ _Float16 ldsWA[16 * DV];   // 8 KB add_W tile
  __shared__ int      idxS[64];
  const int tid = threadIdx.x, lane = tid & 31, w = tid >> 5;
  const long row0 = (long)blockIdx.x * 64;

  if (tid < 64) idxS[tid] = qa_data[row0 + tid];
  __syncthreads();
  for (int i = tid; i < 64 * DV; i += 128) {
    const int r = i >> 8, c = i & 255;
    ldsQA[i] = (_Float16)qa_embed[(long)idxS[r] * DV + c];
  }
  __syncthreads();

  // Cache full-K A fragments in registers, reuse for all 16 N-tiles.
  v16h af[8];
#pragma unroll
  for (int k = 0; k < 8; ++k) af[k] = frag16(ldsQA + w * 16 * DV, DV, lane, k * 32);

  for (int nt = 0; nt < 16; ++nt) {   // N = 256 -> 16 tiles
    __syncthreads();
    for (int i = tid; i < 16 * DV; i += 128) {
      const int r = i >> 8, c = i & 255;
      ldsWE[i] = (_Float16)erase_W[(nt * 16 + r) * DV + c];
      ldsWA[i] = (_Float16)add_W[(nt * 16 + r) * DV + c];
    }
    __syncthreads();
    if (nt + 1 < 16) __builtin_prefetch(&erase_W[(nt + 1) * 16 * DV], 0, 0);

    v8f accE = {0.f, 0.f, 0.f, 0.f, 0.f, 0.f, 0.f, 0.f};
    v8f accA = {0.f, 0.f, 0.f, 0.f, 0.f, 0.f, 0.f, 0.f};
#pragma unroll
    for (int k = 0; k < 8; ++k) {     // K = 256 -> 8 k-steps
      accE = wmma16(af[k], frag16(ldsWE, DV, lane, k * 32), accE);
      accA = wmma16(af[k], frag16(ldsWA, DV, lane, k * 32), accA);
    }
    const int n = nt * 16 + (lane & 15);
    const float eb = erase_b[n], ab = add_b[n];
    const long rbase = row0 + w * 16 + 8 * (lane >> 4);
#pragma unroll
    for (int r = 0; r < 8; ++r) {
      eraseO[(rbase + r) * DV + n] = (_Float16)sigf(accE[r] + eb);
      addO[(rbase + r) * DV + n]   = (_Float16)tanhf(accA[r] + ab);
    }
  }
}

// ---- Stage 3: sequential memory scan. One WG per batch; each thread owns
// one DV column; the 64-slot memory column lives entirely in 64 VGPRs.
__global__ __launch_bounds__(256)
void scan_kernel(const float* __restrict__ init_mem, const float* __restrict__ cw,
                 const _Float16* __restrict__ eraseO, const _Float16* __restrict__ addO,
                 _Float16* __restrict__ readsO) {
  __shared__ float wS[MEM];
  const int d = threadIdx.x;        // 0..255 : DV column
  const int b = blockIdx.x;         // 0..511 : batch
  float mem[MEM];
#pragma unroll
  for (int m = 0; m < MEM; ++m) mem[m] = init_mem[m * DV + d];

  for (int s = 0; s < SEQLEN; ++s) {
    const long row = (long)b * SEQLEN + s;
    __syncthreads();
    if (d < MEM) wS[d] = cw[row * MEM + d];
    __syncthreads();
    const float er = (float)eraseO[row * DV + d];
    const float ad = (float)addO[row * DV + d];
    float rd = 0.f;
#pragma unroll
    for (int m = 0; m < MEM; ++m) {
      const float wm = wS[m];
      rd += wm * mem[m];                              // read (pre-update)
      mem[m] = mem[m] * (1.f - wm * er) + wm * ad;    // erase/add update
    }
    readsO[row * DV + d] = (_Float16)rd;
  }
}

// ---- Stage 4: h = tanh([reads|q_e] @ read_W^T + b); logit; BCE + outputs.
__global__ __launch_bounds__(128)
void fc_kernel(const int* __restrict__ q_data, const float* __restrict__ q_embed,
               const _Float16* __restrict__ readsO,
               const float* __restrict__ read_W, const float* __restrict__ read_b,
               const float* __restrict__ pred_W, const float* __restrict__ pred_b,
               const float* __restrict__ target, float* __restrict__ out,
               float* __restrict__ accum) {
  __shared__ _Float16 ldsW[16 * KIN];   // 12 KB read_W tile (f16)
  __shared__ _Float16 ldsH[64 * FCN];   // 16 KB h activations (f16)
  __shared__ int      idxS[64];
  __shared__ float    red[128], redc[128];
  const int tid = threadIdx.x, lane = tid & 31, w = tid >> 5;
  const long row0 = (long)blockIdx.x * 64;

  if (tid < 64) idxS[tid] = q_data[row0 + tid];
  __syncthreads();

  // A fragments straight from global: K 0..255 = reads (f16), 256..383 = q_e (f32->f16)
  const int M = lane & 15;
  const int hi = (lane >> 4) & 1;
  const long grow = row0 + w * 16 + M;
  const _Float16* rrow = readsO + grow * DV;
  const float* qrow = q_embed + (long)idxS[w * 16 + M] * DK;
  v16h af[12];
#pragma unroll
  for (int k = 0; k < 12; ++k) {
    v16h a;
#pragma unroll
    for (int j = 0; j < 8; ++j) {
      const int off = ((2 * j) & 7) + 8 * hi + 16 * (j >> 2);
      if (k < 8) {
        const int K = k * 32 + off;
        a[2 * j] = rrow[K];  a[2 * j + 1] = rrow[K + 1];
      } else {
        const int K = (k - 8) * 32 + off;
        a[2 * j] = (_Float16)qrow[K];  a[2 * j + 1] = (_Float16)qrow[K + 1];
      }
    }
    af[k] = a;
  }

  for (int nt = 0; nt < 8; ++nt) {    // N = 128 -> 8 tiles
    __syncthreads();
    for (int i = tid; i < 16 * KIN; i += 128) {
      const int r = i / KIN, c = i - r * KIN;
      ldsW[i] = (_Float16)read_W[(nt * 16 + r) * KIN + c];
    }
    __syncthreads();
    v8f acc = {0.f, 0.f, 0.f, 0.f, 0.f, 0.f, 0.f, 0.f};
#pragma unroll
    for (int k = 0; k < 12; ++k)      // K = 384 -> 12 k-steps
      acc = wmma16(af[k], frag16(ldsW, KIN, lane, k * 32), acc);
    const int n = nt * 16 + (lane & 15);
    const float rb = read_b[n];
    const int rbase = w * 16 + 8 * (lane >> 4);
#pragma unroll
    for (int r = 0; r < 8; ++r)
      ldsH[(rbase + r) * FCN + n] = (_Float16)tanhf(acc[r] + rb);
  }
  __syncthreads();

  float bsum = 0.f, bcnt = 0.f;
  if (tid < 64) {
    float logit = pred_b[0];
#pragma unroll 16
    for (int f = 0; f < FCN; ++f) logit += (float)ldsH[tid * FCN + f] * pred_W[f];
    const long rowg = row0 + tid;
    const float t = target[rowg];
    const bool mask = t >= 0.f;
    const float pred = sigf(logit);
    const float bce = fmaxf(logit, 0.f) - logit * t + log1pf(expf(-fabsf(logit)));
    out[1 + rowg] = mask ? pred : 0.f;
    out[1 + (long)ROWS + rowg] = mask ? t : 0.f;
    if (mask) { bsum = bce; bcnt = 1.f; }
  }
  red[tid] = bsum; redc[tid] = bcnt;
  __syncthreads();
  for (int s = 64; s > 0; s >>= 1) {
    if (tid < s) { red[tid] += red[tid + s]; redc[tid] += redc[tid + s]; }
    __syncthreads();
  }
  if (tid == 0) { atomicAdd(&accum[0], red[0]); atomicAdd(&accum[1], redc[0]); }
}

__global__ void zero_kernel(float* accum) { accum[0] = 0.f; accum[1] = 0.f; }
__global__ void fin_kernel(const float* accum, float* out) {
  out[0] = accum[0] / fmaxf(accum[1], 1.f);
}

// ---- Host launch ------------------------------------------------------------
extern "C" void kernel_launch(void* const* d_in, const int* in_sizes, int n_in,
                              void* d_out, int out_size, void* d_ws, size_t ws_size,
                              hipStream_t stream) {
  const int*   q_data   = (const int*)d_in[0];
  const int*   qa_data  = (const int*)d_in[1];
  const float* target   = (const float*)d_in[2];
  const float* q_embed  = (const float*)d_in[3];
  const float* qa_embed = (const float*)d_in[4];
  const float* key_mem  = (const float*)d_in[5];
  const float* init_mem = (const float*)d_in[6];
  const float* erase_W  = (const float*)d_in[7];
  const float* erase_b  = (const float*)d_in[8];
  const float* add_W    = (const float*)d_in[9];
  const float* add_b    = (const float*)d_in[10];
  const float* read_W   = (const float*)d_in[11];
  const float* read_b   = (const float*)d_in[12];
  const float* pred_W   = (const float*)d_in[13];
  const float* pred_b   = (const float*)d_in[14];
  float* out = (float*)d_out;

  char* ws = (char*)d_ws;
  size_t off = 0;
  float*    cw     = (float*)(ws + off);    off += (size_t)ROWS * MEM * sizeof(float);
  _Float16* eraseO = (_Float16*)(ws + off); off += (size_t)ROWS * DV * sizeof(_Float16);
  _Float16* addO   = (_Float16*)(ws + off); off += (size_t)ROWS * DV * sizeof(_Float16);
  _Float16* readsO = (_Float16*)(ws + off); off += (size_t)ROWS * DV * sizeof(_Float16);
  float*    accum  = (float*)(ws + off);    off += 2 * sizeof(float);

  zero_kernel<<<1, 1, 0, stream>>>(accum);
  cw_kernel<<<ROWS / 64, 128, 0, stream>>>(q_data, q_embed, key_mem, cw);
  ea_kernel<<<ROWS / 64, 128, 0, stream>>>(qa_data, qa_embed, erase_W, erase_b,
                                           add_W, add_b, eraseO, addO);
  scan_kernel<<<BATCH, 256, 0, stream>>>(init_mem, cw, eraseO, addO, readsO);
  fc_kernel<<<ROWS / 64, 128, 0, stream>>>(q_data, q_embed, readsO, read_W, read_b,
                                           pred_W, pred_b, target, out, accum);
  fin_kernel<<<1, 1, 0, stream>>>(accum, out);
}
